// STGNNMixer_7086696038676
// MI455X (gfx1250) — matrix operand order repaired and checked by reading, not compiled, not run on hardware
//
#include <hip/hip_runtime.h>

// ---------------- problem constants ----------------
#define Bq    4
#define NNODE 1437
#define SSZ   56
#define PPZ   28
#define FINZ  10
#define FFU   6
#define HHZ   64
#define NPAD  1440            // N padded to /16 for WMMA tiles
#define DDZ   3584            // S*H
#define NKC   45              // NPAD/32 k-chunks
#define NTN   90              // NPAD/16 row tiles
#define NDT   224             // DDZ/16 col tiles
#define BNSH  (Bq*NNODE*SSZ*HHZ)   // 20,600,832
#define NROWS (Bq*NNODE*SSZ)       // 321,888

typedef __attribute__((ext_vector_type(16))) _Float16 v16h;
typedef __attribute__((ext_vector_type(8)))  _Float16 v8h;
typedef __attribute__((ext_vector_type(8)))  float    v8f;

__device__ __forceinline__ float gelu_f(float x) {
  return 0.5f * x * (1.0f + erff(x * 0.70710678118654752440f));
}

// block-wide sum for 64-thread blocks
__device__ __forceinline__ float bsum64(float v, float* sm) {
  int t = threadIdx.x;
  sm[t] = v; __syncthreads();
  if (t < 32) sm[t] += sm[t + 32]; __syncthreads();
  if (t < 16) sm[t] += sm[t + 16]; __syncthreads();
  if (t <  8) sm[t] += sm[t +  8]; __syncthreads();
  if (t <  4) sm[t] += sm[t +  4]; __syncthreads();
  if (t <  2) sm[t] += sm[t +  2]; __syncthreads();
  if (t <  1) sm[t] += sm[t +  1]; __syncthreads();
  float r = sm[0]; __syncthreads();
  return r;
}

// ---- 1. input projection + LayerNorm + normalized node embedding add ----
__global__ void proj_ln_emb_kernel(const float* __restrict__ x,
                                   const float* __restrict__ fp_w,
                                   const float* __restrict__ fp_b,
                                   const float* __restrict__ lg,
                                   const float* __restrict__ lb,
                                   const float* __restrict__ emb,
                                   float* __restrict__ h) {
  __shared__ float sm[64];
  size_t row = blockIdx.x;          // (b*N + n)*S + s
  int t = threadIdx.x;
  const float* xr = x + row * FINZ;
  float acc = fp_b[t];
#pragma unroll
  for (int f = 0; f < FINZ; ++f) acc += xr[f] * fp_w[t * FINZ + f];
  float mean = bsum64(acc, sm) * (1.0f / 64.0f);
  float d = acc - mean;
  float var = bsum64(d * d, sm) * (1.0f / 64.0f);
  float v = d * rsqrtf(var + 1e-5f) * lg[t] + lb[t];
  size_t n = (row / SSZ) % NNODE;
  float e = emb[n * HHZ + t];
  float nrm2 = bsum64(e * e, sm);
  v += e / fmaxf(sqrtf(nrm2), 1e-12f);
  h[row * HHZ + t] = v;
}

// ---- generic per-row (H=64) LayerNorm ----
__global__ void ln_kernel(const float* __restrict__ in, float* __restrict__ out,
                          const float* __restrict__ g, const float* __restrict__ b) {
  __shared__ float sm[64];
  size_t row = blockIdx.x;
  int t = threadIdx.x;
  float v = in[row * HHZ + t];
  float mean = bsum64(v, sm) * (1.0f / 64.0f);
  float d = v - mean;
  float var = bsum64(d * d, sm) * (1.0f / 64.0f);
  out[row * HHZ + t] = d * rsqrtf(var + 1e-5f) * g[t] + b[t];
}

// ---- 2a. recency-weighted context ----
__global__ void ctx_kernel(const float* __restrict__ h, float* __restrict__ ctx) {
  size_t bn = blockIdx.x;
  int t = threadIdx.x;
  float Z = 0.0f, acc = 0.0f;
  for (int s = 0; s < SSZ; ++s) {
    float w = expf(-(float)(SSZ - s) / 14.0f);
    Z += w;
    acc += w * h[(bn * SSZ + s) * HHZ + t];
  }
  ctx[bn * HHZ + t] = acc / Z;
}

// ---- 2b. Q/K projections -> zero-padded f16 [B,1440,64] ----
__global__ void qk_kernel(const float* __restrict__ ctx,
                          const float* __restrict__ qw, const float* __restrict__ qb,
                          const float* __restrict__ kw, const float* __restrict__ kb,
                          _Float16* __restrict__ Qh, _Float16* __restrict__ Kh) {
  int b = blockIdx.y, n = blockIdx.x, t = threadIdx.x;
  size_t orow = ((size_t)b * NPAD + n) * HHZ;
  if (n >= NNODE) { Qh[orow + t] = (_Float16)0.0f; Kh[orow + t] = (_Float16)0.0f; return; }
  __shared__ float c[64];
  c[t] = ctx[((size_t)b * NNODE + n) * HHZ + t];
  __syncthreads();
  float q = qb[t], k = kb[t];
#pragma unroll 8
  for (int j = 0; j < HHZ; ++j) { q += qw[t * HHZ + j] * c[j]; k += kw[t * HHZ + j] * c[j]; }
  Qh[orow + t] = (_Float16)q; Kh[orow + t] = (_Float16)k;
}

// ---- 3. da = relu(clip(Q K^T / 8)) via WMMA; 1 wave per 16x16 tile ----
__global__ void da_kernel(const _Float16* __restrict__ Qh,
                          const _Float16* __restrict__ Kh,
                          float* __restrict__ da) {
  int tile = blockIdx.x;
  int tm = tile / NTN, tn = tile % NTN;
  int b = blockIdx.y;
  int lane = threadIdx.x, lo = lane & 15, hi = lane >> 4;
  const _Float16* Qrow = Qh + ((size_t)b * NPAD + tm * 16 + lo) * HHZ;
  const _Float16* Krow = Kh + ((size_t)b * NPAD + tn * 16 + lo) * HHZ;
  v8f acc = {};
#pragma unroll
  for (int k0 = 0; k0 < HHZ; k0 += 32) {
    v16h a, bb;
    v8h a0 = *(const v8h*)(Qrow + k0 + hi * 8);        // A: interleaved K layout
    v8h a1 = *(const v8h*)(Qrow + k0 + 16 + hi * 8);
    v8h b0 = *(const v8h*)(Krow + k0 + hi * 8);        // B: interleaved K layout
    v8h b1 = *(const v8h*)(Krow + k0 + 16 + hi * 8);
#pragma unroll
    for (int j = 0; j < 8; ++j) {
      a[j] = a0[j]; a[j + 8] = a1[j];
      bb[j] = b0[j]; bb[j + 8] = b1[j];
    }
    acc = __builtin_amdgcn_wmma_f32_16x16x32_f16(false, a, false, bb,
                                                 (short)0, acc, false, false);
  }
#pragma unroll
  for (int r = 0; r < 8; ++r) {
    int m = tm * 16 + hi * 8 + r;     // Q row
    int n = tn * 16 + lo;             // K row (column of da)
    if (m < NNODE && n < NNODE) {
      float v = acc[r] * 0.125f;                // / sqrt(64)
      v = fminf(fmaxf(v, 0.0f), 10.0f);         // clip(-10,10) then relu
      da[((size_t)b * NNODE + m) * NNODE + n] = v;
    }
  }
}

// WMMA-A fragment address for matrix element (row tile tn, row-lane lo, column k).
// Layout: [tn][kc][lane][j], lane = hi*16+lo, k = kc*32 + (j>>3)*16 + hi*8 + (j&7)
__device__ __forceinline__ size_t afrag_addr(size_t tilebase, int kcol, int lo) {
  int kc = kcol >> 5, r = kcol & 31;
  int lane = (((r >> 3) & 1) << 4) + lo;
  int j = ((r >> 4) << 3) + (r & 7);
  return tilebase + (size_t)kc * 512 + lane * 16 + j;
}

// ---- 4. top-5 + softmax + alpha-mix with adj + L1 normalize ->
//         f16 comb in WMMA-A fragment layout [B][90][45][512]
__global__ void comb_kernel(const float* __restrict__ da,
                            const float* __restrict__ adj,
                            const float* __restrict__ ga,
                            _Float16* __restrict__ combf) {
  int b = blockIdx.y, n = blockIdx.x, t = threadIdx.x;
  int tnn = n >> 4, lo = n & 15;
  size_t tilebase = (((size_t)b * NTN + tnn) * NKC) * 512;
  if (n >= NNODE) {
    for (int m = t; m < NPAD; m += 256)
      combf[afrag_addr(tilebase, m, lo)] = (_Float16)0.0f;
    return;
  }
  __shared__ float vals[NNODE];
  __shared__ float red[256];
  __shared__ int   redi[256];
  __shared__ int   topi[5];
  __shared__ float topv[5];
  __shared__ float sdenom, salpha;
  const float* drow = da + ((size_t)b * NNODE + n) * NNODE;
  for (int m = t; m < NNODE; m += 256) vals[m] = drow[m];
  __syncthreads();
  for (int kk = 0; kk < 5; ++kk) {               // sequential argmax x5
    float bv = -1.0f; int bi = NNODE;
    for (int m = t; m < NNODE; m += 256) {
      float v = vals[m];
      if (v > bv || (v == bv && m < bi)) { bv = v; bi = m; }
    }
    red[t] = bv; redi[t] = bi; __syncthreads();
    for (int off = 128; off > 0; off >>= 1) {
      if (t < off) {
        float ov = red[t + off]; int oi = redi[t + off];
        if (ov > red[t] || (ov == red[t] && oi < redi[t])) { red[t] = ov; redi[t] = oi; }
      }
      __syncthreads();
    }
    if (t == 0) { topi[kk] = redi[0]; topv[kk] = red[0]; vals[redi[0]] = -1.0f; }
    __syncthreads();
  }
  float s = 0.0f;                                 // sum |adj| over row n
  for (int m = t; m < NNODE; m += 256) s += fabsf(adj[(size_t)n * NNODE + m]);
  red[t] = s; __syncthreads();
  for (int off = 128; off > 0; off >>= 1) { if (t < off) red[t] += red[t + off]; __syncthreads(); }
  if (t == 0) {
    float alpha = 1.0f / (1.0f + expf(-ga[0]));
    float mx = topv[0];
    float p[5], se = 0.0f;
    for (int kk = 0; kk < 5; ++kk) { p[kk] = expf(topv[kk] - mx); se += p[kk]; }
    for (int kk = 0; kk < 5; ++kk) topv[kk] = p[kk] / se;    // dyn softmax probs
    sdenom = fmaxf(alpha * red[0] + (1.0f - alpha), 1e-12f); // |comb| row sum
    salpha = alpha;
  }
  __syncthreads();
  float alpha = salpha, inv = 1.0f / sdenom;
  for (int m = t; m < NPAD; m += 256) {
    float c = 0.0f;
    if (m < NNODE) {
      c = alpha * adj[(size_t)n * NNODE + m];
      for (int kk = 0; kk < 5; ++kk) if (topi[kk] == m) c += (1.0f - alpha) * topv[kk];
      c *= inv;
    }
    combf[afrag_addr(tilebase, m, lo)] = (_Float16)c;
  }
}

// ---- 5a. depthwise weekly(7)/monthly(28) "same" convs, residual add ----
__global__ void conv_kernel(const float* __restrict__ xn,
                            const float* __restrict__ wk,
                            const float* __restrict__ mk,
                            float* __restrict__ h, int total) {
  int i = blockIdx.x * 256 + threadIdx.x;
  if (i >= total) return;
  int f = i & 63;
  int s = (i >> 6) % SSZ;
  size_t bn = (size_t)i / (SSZ * HHZ);
  const float* base = xn + bn * SSZ * HHZ + f;
  float acc = 0.0f;
#pragma unroll
  for (int t2 = 0; t2 < 7; ++t2) {
    int s2 = s - 3 + t2;
    if (s2 >= 0 && s2 < SSZ) acc += base[s2 * HHZ] * wk[t2];
  }
#pragma unroll
  for (int t2 = 0; t2 < 28; ++t2) {
    int s2 = s - 13 + t2;
    if (s2 >= 0 && s2 < SSZ) acc += base[s2 * HHZ] * mk[t2];
  }
  h[i] += acc;
}

// ---- 5b. temporal MLP over S (GELU), residual add ----
__global__ void temporal_kernel(const float* __restrict__ xn,
                                const float* __restrict__ tw,
                                const float* __restrict__ tb,
                                float* __restrict__ h, int total) {
  int i = blockIdx.x * 256 + threadIdx.x;
  if (i >= total) return;
  int so = i % SSZ;
  int f = (i / SSZ) & 63;
  size_t bn = (size_t)i / (SSZ * HHZ);
  const float* base = xn + bn * SSZ * HHZ + f;
  float acc = tb[so];
#pragma unroll 8
  for (int s = 0; s < SSZ; ++s) acc += base[s * HHZ] * tw[so * SSZ + s];
  h[(bn * SSZ + so) * HHZ + f] += gelu_f(acc);
}

// ---- 5c. feature MLP over H (GELU), residual add ----
__global__ void feature_kernel(const float* __restrict__ xn,
                               const float* __restrict__ fw,
                               const float* __restrict__ fb,
                               float* __restrict__ h, int total) {
  int i = blockIdx.x * 256 + threadIdx.x;
  if (i >= total) return;
  int o = i & 63;
  size_t row = (size_t)i >> 6;
  const float* r = xn + row * HHZ;
  float acc = fb[o];
#pragma unroll 8
  for (int f = 0; f < HHZ; ++f) acc += r[f] * fw[o * HHZ + f];
  h[i] += gelu_f(acc);
}

// ---- 5d. spatial projection -> f16 xp in WMMA-B fragment layout [B][224][45][512]
//      element (k=n, d): lane = ((k%32>>3)&1)*16 + (d&15), elem = (k%32>>4)*8 + (k&7)
__global__ void spatproj_kernel(const float* __restrict__ xn,
                                const float* __restrict__ sw,
                                _Float16* __restrict__ xpf, int total) {
  int i = blockIdx.x * 256 + threadIdx.x;
  if (i >= total) return;
  int o = i & 63;
  int s = (i >> 6) % SSZ;
  int n = (i / (SSZ * HHZ)) % NPAD;
  int b = i / (SSZ * HHZ * NPAD);
  float acc = 0.0f;
  if (n < NNODE) {
    const float* r = xn + (((size_t)b * NNODE + n) * SSZ + s) * HHZ;
#pragma unroll 8
    for (int f = 0; f < HHZ; ++f) acc += r[f] * sw[o * HHZ + f];
  }
  int d = s * HHZ + o;
  int dt = d >> 4, lo = d & 15;
  int kc = n >> 5, rr = n & 31;
  int lane = (((rr >> 3) & 1) << 4) + lo;
  int e = ((rr >> 4) << 3) + (rr & 7);
  xpf[(((size_t)b * NDT + dt) * NKC + kc) * 512 + lane * 16 + e] = (_Float16)acc;
}

// ---- 5e. spatial bmm: h += comb[b] x xp[b] + sb ----
// LDS-free: both operands pre-swizzled to fragment layout, so every lane reads
// a contiguous 32B fragment (wave = 1KB coalesced block per tile per k-chunk).
// Steady state: 10 global_load_b128 + 4 v_wmma per 32-wide k-chunk, all L2-hit.
__global__ void bmm_kernel(const _Float16* __restrict__ Af,  // [B][90][45][512]
                           const _Float16* __restrict__ Bf,  // [B][224][45][512]
                           const float* __restrict__ sb,
                           float* __restrict__ h) {
  const int dg = blockIdx.x;    // 0..55  (4 consecutive d-tiles)
  const int tn = blockIdx.y;    // 0..89  (16 output rows)
  const int b  = blockIdx.z;
  const int lane = threadIdx.x, lo = lane & 15, hi = lane >> 4;
  const _Float16* Abase = Af + (((size_t)b * NTN + tn) * NKC) * 512 + lane * 16;
  const _Float16* Bbase = Bf + (((size_t)b * NDT + dg * 4) * NKC) * 512 + lane * 16;
  v8f acc[4] = {};
  for (int kc = 0; kc < NKC; ++kc) {
    const _Float16* ap = Abase + (size_t)kc * 512;
    if (kc + 1 < NKC) __builtin_prefetch(ap + 512, 0, 1);   // global_prefetch_b8
    v16h a;
    {
      v8h a0 = *(const v8h*)(ap);
      v8h a1 = *(const v8h*)(ap + 8);
#pragma unroll
      for (int j = 0; j < 8; ++j) { a[j] = a0[j]; a[j + 8] = a1[j]; }
    }
#pragma unroll
    for (int t = 0; t < 4; ++t) {
      const _Float16* bp = Bbase + ((size_t)t * NKC + kc) * 512;
      if (kc + 1 < NKC) __builtin_prefetch(bp + 512, 0, 1);
      v16h bb;
      v8h b0 = *(const v8h*)(bp);
      v8h b1 = *(const v8h*)(bp + 8);
#pragma unroll
      for (int j = 0; j < 8; ++j) { bb[j] = b0[j]; bb[j + 8] = b1[j]; }
      acc[t] = __builtin_amdgcn_wmma_f32_16x16x32_f16(false, a, false, bb,
                                                      (short)0, acc[t], false, false);
    }
  }
#pragma unroll
  for (int t = 0; t < 4; ++t) {
#pragma unroll
    for (int r = 0; r < 8; ++r) {
      int nn = tn * 16 + hi * 8 + r;
      if (nn < NNODE) {
        int d = (dg * 4 + t) * 16 + lo;
        h[((size_t)b * NNODE + nn) * DDZ + d] += acc[t][r] + sb[d & 63];
      }
    }
  }
}

// ---- 6a. future projection ----
__global__ void futproj_kernel(const float* __restrict__ xf,
                               const float* __restrict__ fw,
                               const float* __restrict__ fb,
                               float* __restrict__ fut, int total) {
  int i = blockIdx.x * 256 + threadIdx.x;
  if (i >= total) return;
  int hh = i & 63;
  size_t row = (size_t)i >> 6;     // b*N*P rows
  const float* r = xf + row * FFU;
  float acc = fb[hh];
#pragma unroll
  for (int f = 0; f < FFU; ++f) acc += r[f] * fw[hh * FFU + f];
  fut[i] = acc;
}

// ---- 6b. head: out[b,n,p] = concat(hist,fut) . head_w[p] + head_b[p] ----
__global__ void head_kernel(const float* __restrict__ h,
                            const float* __restrict__ fut,
                            const float* __restrict__ hw,
                            const float* __restrict__ hb,
                            float* __restrict__ out, int total) {
  int i = blockIdx.x * 256 + threadIdx.x;
  if (i >= total) return;
  int p = i % PPZ;
  size_t bn = (size_t)i / PPZ;
  const float* w = hw + (size_t)p * (DDZ + PPZ * HHZ);
  const float* hr = h + bn * DDZ;
  const float* fr = fut + bn * (PPZ * HHZ);
  float acc = hb[p];
  for (int d = 0; d < DDZ; ++d) acc += hr[d] * w[d];
  for (int d = 0; d < PPZ * HHZ; ++d) acc += fr[d] * w[DDZ + d];
  out[i] = acc;
}

extern "C" void kernel_launch(void* const* d_in, const int* in_sizes, int n_in,
                              void* d_out, int out_size, void* d_ws, size_t ws_size,
                              hipStream_t stream) {
  (void)in_sizes; (void)n_in; (void)out_size; (void)ws_size;
  const float* x         = (const float*)d_in[0];
  const float* x_future  = (const float*)d_in[1];
  const float* adj       = (const float*)d_in[2];
  const float* fp_w      = (const float*)d_in[3];
  const float* fp_b      = (const float*)d_in[4];
  const float* in_ln_g   = (const float*)d_in[5];
  const float* in_ln_b   = (const float*)d_in[6];
  const float* q_w       = (const float*)d_in[7];
  const float* q_b       = (const float*)d_in[8];
  const float* k_w       = (const float*)d_in[9];
  const float* k_b       = (const float*)d_in[10];
  const float* node_emb  = (const float*)d_in[11];
  const float* graph_alpha = (const float*)d_in[12];
  const float* blk_weekly  = (const float*)d_in[13];
  const float* blk_monthly = (const float*)d_in[14];
  const float* lnc_g = (const float*)d_in[15];
  const float* lnc_b = (const float*)d_in[16];
  const float* lnt_g = (const float*)d_in[17];
  const float* lnt_b = (const float*)d_in[18];
  const float* lnf_g = (const float*)d_in[19];
  const float* lnf_b = (const float*)d_in[20];
  const float* lns_g = (const float*)d_in[21];
  const float* lns_b = (const float*)d_in[22];
  const float* temp_w = (const float*)d_in[23];
  const float* temp_b = (const float*)d_in[24];
  const float* feat_w = (const float*)d_in[25];
  const float* feat_b = (const float*)d_in[26];
  const float* spat_w = (const float*)d_in[27];
  const float* spat_b = (const float*)d_in[28];
  const float* futr_w = (const float*)d_in[29];
  const float* futr_b = (const float*)d_in[30];
  const float* head_w = (const float*)d_in[31];
  const float* head_b = (const float*)d_in[32];
  float* out = (float*)d_out;

  // ---- workspace carving (~300 MB) ----
  char* wp = (char*)d_ws;
  auto carve = [&](size_t bytes) -> char* {
    char* p = wp; wp += (bytes + 255) & ~(size_t)255; return p;
  };
  float*    hbuf  = (float*)carve((size_t)BNSH * 4);
  float*    tmp   = (float*)carve((size_t)BNSH * 4);
  float*    ctx   = (float*)carve((size_t)Bq * NNODE * HHZ * 4);
  _Float16* Qh    = (_Float16*)carve((size_t)Bq * NPAD * HHZ * 2);
  _Float16* Kh    = (_Float16*)carve((size_t)Bq * NPAD * HHZ * 2);
  float*    da    = (float*)carve((size_t)Bq * NNODE * NNODE * 4);
  _Float16* combf = (_Float16*)carve((size_t)Bq * NTN * NKC * 512 * 2);
  _Float16* xpf   = (_Float16*)carve((size_t)Bq * NDT * NKC * 512 * 2);
  float*    fut   = (float*)carve((size_t)Bq * NNODE * PPZ * HHZ * 4);

  // ---- stem ----
  proj_ln_emb_kernel<<<NROWS, 64, 0, stream>>>(x, fp_w, fp_b, in_ln_g, in_ln_b,
                                               node_emb, hbuf);
  ctx_kernel<<<Bq * NNODE, 64, 0, stream>>>(hbuf, ctx);
  qk_kernel<<<dim3(NPAD, Bq), 64, 0, stream>>>(ctx, q_w, q_b, k_w, k_b, Qh, Kh);
  da_kernel<<<dim3(NTN * NTN, Bq), 32, 0, stream>>>(Qh, Kh, da);
  comb_kernel<<<dim3(NPAD, Bq), 256, 0, stream>>>(da, adj, graph_alpha, combf);

  // ---- mixer blocks ----
  for (int i = 0; i < 2; ++i) {
    const int totalE = BNSH;
    ln_kernel<<<NROWS, 64, 0, stream>>>(hbuf, tmp, lnc_g + i * HHZ, lnc_b + i * HHZ);
    conv_kernel<<<(totalE + 255) / 256, 256, 0, stream>>>(
        tmp, blk_weekly + i * 7, blk_monthly + i * 28, hbuf, totalE);
    ln_kernel<<<NROWS, 64, 0, stream>>>(hbuf, tmp, lnt_g + i * HHZ, lnt_b + i * HHZ);
    temporal_kernel<<<(totalE + 255) / 256, 256, 0, stream>>>(
        tmp, temp_w + i * SSZ * SSZ, temp_b + i * SSZ, hbuf, totalE);
    ln_kernel<<<NROWS, 64, 0, stream>>>(hbuf, tmp, lnf_g + i * HHZ, lnf_b + i * HHZ);
    feature_kernel<<<(totalE + 255) / 256, 256, 0, stream>>>(
        tmp, feat_w + i * HHZ * HHZ, feat_b + i * HHZ, hbuf, totalE);
    ln_kernel<<<NROWS, 64, 0, stream>>>(hbuf, tmp, lns_g + i * HHZ, lns_b + i * HHZ);
    const int totalP = Bq * NPAD * SSZ * HHZ;
    spatproj_kernel<<<(totalP + 255) / 256, 256, 0, stream>>>(
        tmp, spat_w + i * HHZ * HHZ, xpf, totalP);
    bmm_kernel<<<dim3(DDZ / 64, NTN, Bq), 32, 0, stream>>>(
        combf, xpf, spat_b + i * HHZ, hbuf);
  }

  // ---- head ----
  const int totalF = Bq * NNODE * PPZ * HHZ;
  futproj_kernel<<<(totalF + 255) / 256, 256, 0, stream>>>(x_future, futr_w, futr_b,
                                                           fut, totalF);
  const int totalO = Bq * NNODE * PPZ;
  head_kernel<<<(totalO + 255) / 256, 256, 0, stream>>>(hbuf, fut, head_w, head_b,
                                                        out, totalO);
}
